// GATLayer_79199196938866
// MI455X (gfx1250) — compile-verified, matrix operand order checked by reading, not run on hardware
//
#include <hip/hip_runtime.h>

// GAT layer for MI455X (gfx1250, wave32).
// N=4096 nodes, IN=512, H=8 heads, D=64.
// bf16 WMMA (f32 accumulate) for both GEMMs; flash-style fused masked-softmax
// attention (never materializes the 512MB attention tensor). Softmax state in
// fp32 registers; adj tiles staged once per 8 heads through LDS — with
// double-buffered ASYNCcnt-tracked global_load_async_to_lds when available.

#define N_NODES 4096
#define IN_F    512
#define NH      8
#define HD      64
#define NF      512          // NH*HD
#define ALPHA   0.2f
#define NEG_BIG (-9000000000000000.0f)
#define ADJ_STRIDE 36        // LDS row stride (ints): conflict-free + 16B aligned

#if __has_builtin(__builtin_amdgcn_global_load_async_to_lds_b32) && \
    __has_builtin(__builtin_amdgcn_s_wait_asynccnt)
#define USE_ASYNC_LDS 1
#else
#define USE_ASYNC_LDS 0
#endif

typedef __bf16 v16bf __attribute__((ext_vector_type(16)));
typedef __bf16 v8bf  __attribute__((ext_vector_type(8)));
typedef float  v8f   __attribute__((ext_vector_type(8)));
typedef float  v4f   __attribute__((ext_vector_type(4)));
typedef int    v4i   __attribute__((ext_vector_type(4)));

static __device__ __forceinline__ __bf16 f2bf(float f) {
    // round-to-nearest-even fp32 -> bf16
    unsigned u = __builtin_bit_cast(unsigned, f);
    unsigned r = (u + 0x7FFFu + ((u >> 16) & 1u)) >> 16;
    return __builtin_bit_cast(__bf16, (unsigned short)r);
}

// Load a 16-element bf16 fragment whose ISA layout is two contiguous 8-elem
// runs at p[0..7] and p[16..23] (wave32 16-bit A/B K-grouping).
static __device__ __forceinline__ v16bf ld_bfrag(const __bf16* p) {
    v8bf lo = *(const v8bf*)p;
    v8bf hi = *(const v8bf*)(p + 16);
    return __builtin_shufflevector(lo, hi,
        0, 1, 2, 3, 4, 5, 6, 7, 8, 9, 10, 11, 12, 13, 14, 15);
}

#if USE_ASYNC_LDS
typedef __attribute__((address_space(1))) int gas_int;   // global
typedef __attribute__((address_space(3))) int las_int;   // LDS
// ASYNCcnt-tracked direct global->LDS copy (no VGPR round trip).
static __device__ __forceinline__ void async_ld_b32(const int* g, int* l) {
    __builtin_amdgcn_global_load_async_to_lds_b32(
        (gas_int*)g, (las_int*)l, 0, 0);
}
#endif

// ---------------------------------------------------------------- fp32->bf16
__global__ void __launch_bounds__(256)
gat_cvt_bf16(const float* __restrict__ in, unsigned short* __restrict__ outp, int n) {
    __bf16* o  = (__bf16*)outp;
    int idx    = blockIdx.x * blockDim.x + threadIdx.x;
    int stride = gridDim.x * blockDim.x;
    for (; idx < n; idx += stride) o[idx] = f2bf(in[idx]);
}

// fp32 (R x C, row-major) -> bf16 transposed (C x R): gives K-contiguous B.
__global__ void __launch_bounds__(256)
gat_cvt_bf16_T(const float* __restrict__ in, unsigned short* __restrict__ outp,
               int R, int C) {
    __bf16* o  = (__bf16*)outp;
    int idx    = blockIdx.x * blockDim.x + threadIdx.x;
    int stride = gridDim.x * blockDim.x;
    for (; idx < R * C; idx += stride) {
        int r = idx / C, c = idx % C;
        o[c * R + r] = f2bf(in[idx]);
    }
}

// ------------------------------------------------- Wh = h @ W  (bf16 WMMA)
// One wave computes a 16x64 fp32 tile; K swept 32 at a time. B read from the
// transposed WbT so fragments are contiguous. Emits Wh (fp32, row-major) and
// WhbT (bf16, feature-major: [h*64+d][j]) for the attention GEMM.
__global__ void __launch_bounds__(32)
gat_wh_gemm(const unsigned short* __restrict__ hb_u,
            const unsigned short* __restrict__ wbt_u,
            float* __restrict__ Wh,
            unsigned short* __restrict__ whbt_u) {
    const __bf16* hb   = (const __bf16*)hb_u;
    const __bf16* WbT  = (const __bf16*)wbt_u;
    __bf16*       WhbT = (__bf16*)whbt_u;

    const int lane = threadIdx.x;
    const int row  = lane & 15;   // M (A) / N (B,C) index
    const int half = lane >> 4;   // K-group selector per ISA layout

    const int i0 = blockIdx.x * 16;   // node tile
    const int n0 = blockIdx.y * 64;   // output-feature tile

    v8f acc[4] = {};
    for (int k0 = 0; k0 < IN_F; k0 += 32) {
        const v16bf a = ld_bfrag(hb + (i0 + row) * IN_F + k0 + 8 * half);
        #pragma unroll
        for (int t = 0; t < 4; ++t) {
            const v16bf b =
                ld_bfrag(WbT + (n0 + 16 * t + row) * IN_F + k0 + 8 * half);
            acc[t] = __builtin_amdgcn_wmma_f32_16x16x32_bf16(
                false, a, false, b, (short)0, acc[t], false, false);
        }
    }

    // C/D layout: VGPR r -> M = r + 8*half, N = row.
    #pragma unroll
    for (int t = 0; t < 4; ++t)
        #pragma unroll
        for (int r = 0; r < 8; ++r) {
            const int i = i0 + r + 8 * half;
            const int n = n0 + 16 * t + row;
            const float v = acc[t][r];
            Wh[i * NF + n]              = v;
            WhbT[(long)n * N_NODES + i] = f2bf(v);
        }
}

// --------------------------------------- s1/s2: per-(node,head) 64-dots
// One wave per (n,h); head-major output for contiguous j-streaming later.
__global__ void __launch_bounds__(32)
gat_scores(const float* __restrict__ Wh, const float* __restrict__ a,
           float* __restrict__ s1, float* __restrict__ s2) {
    const int n = blockIdx.x;
    const int h = blockIdx.y;
    const int l = threadIdx.x;

    const float* whr = Wh + n * NF + h * HD;
    const float* a1  = a + h * (2 * HD);
    const float* a2  = a1 + HD;

    float p1 = whr[l] * a1[l] + whr[l + 32] * a1[l + 32];
    float p2 = whr[l] * a2[l] + whr[l + 32] * a2[l + 32];
    #pragma unroll
    for (int off = 16; off > 0; off >>= 1) {
        p1 += __shfl_xor(p1, off, 32);
        p2 += __shfl_xor(p2, off, 32);
    }
    if (l == 0) { s1[h * N_NODES + n] = p1; s2[h * N_NODES + n] = p2; }
}

// ------------------- fused masked-softmax attention + h' GEMM + ELU
// One workgroup = 8 waves = 8 heads for a 16-row node tile. The 16x32 adj
// tile is staged once through LDS (coalesced) and shared by all heads;
// double-buffered via async-to-LDS so the fetch of chunk j+1 overlaps the
// softmax+WMMA compute of chunk j. Online softmax state lives in registers
// (partner lanes via shfl_xor 16); per-row rescale via ds_bpermute shuffles.
__global__ void __launch_bounds__(256)
gat_attn(const float* __restrict__ s1, const float* __restrict__ s2,
         const unsigned short* __restrict__ whbt_u, const int* __restrict__ adj,
         float* __restrict__ out) {
    const __bf16* WhbT = (const __bf16*)whbt_u;
    __shared__ __align__(16) int sm_adj[2][16 * ADJ_STRIDE];

    const int tid  = threadIdx.x;
    const int h    = tid >> 5;         // wave == head
    const int lane = tid & 31;
    const int row  = lane & 15;
    const int half = lane >> 4;
    const int i0   = blockIdx.x * 16;
    const int i    = i0 + row;

    const float  s1v = s1[h * N_NODES + i];
    const float* s2h = s2 + h * N_NODES;

    // cooperative adj staging: thread t handles flat elements t and t+256
    const int cr0 = tid >> 5, cc = tid & 31;      // rows 0..7
    const int cr1 = cr0 + 8;                      // rows 8..15
    const long adr0 = (long)(i0 + cr0) * N_NODES + cc;
    const long adr1 = (long)(i0 + cr1) * N_NODES + cc;
    const int  lofs0 = cr0 * ADJ_STRIDE + cc;
    const int  lofs1 = cr1 * ADJ_STRIDE + cc;

    float m_run = -__builtin_inff();
    float l_run = 0.0f;
    v8f acc[4] = {};

#if USE_ASYNC_LDS
    // preamble: stage chunk 0 into buffer 0
    async_ld_b32(adj + adr0, &sm_adj[0][lofs0]);
    async_ld_b32(adj + adr1, &sm_adj[0][lofs1]);
#endif

    for (int j0 = 0; j0 < N_NODES; j0 += 32) {
        const int buf = (j0 >> 5) & 1;
#if USE_ASYNC_LDS
        if (j0 + 32 < N_NODES) {
            // issue next chunk into the other buffer, then wait for current.
            // async loads complete in order -> asynccnt<=2 means ours landed.
            async_ld_b32(adj + adr0 + j0 + 32, &sm_adj[buf ^ 1][lofs0]);
            async_ld_b32(adj + adr1 + j0 + 32, &sm_adj[buf ^ 1][lofs1]);
            __builtin_amdgcn_s_wait_asynccnt(2);
        } else {
            __builtin_amdgcn_s_wait_asynccnt(0);
        }
#else
        sm_adj[buf][lofs0] = adj[adr0 + j0];
        sm_adj[buf][lofs1] = adj[adr1 + j0];
        if (j0 + 256 < N_NODES) {
            __builtin_prefetch(&adj[adr0 + j0 + 256], 0, 1);  // global_prefetch
            __builtin_prefetch(&adj[adr1 + j0 + 256], 0, 1);
        }
#endif
        __syncthreads();   // all waves' current tile visible in LDS

        // masked leaky-relu logits for this lane's (row, K-group)
        const v4f s2v0 = *(const v4f*)(s2h + j0 + 8 * half);
        const v4f s2v1 = *(const v4f*)(s2h + j0 + 8 * half + 4);
        const v4f s2v2 = *(const v4f*)(s2h + j0 + 16 + 8 * half);
        const v4f s2v3 = *(const v4f*)(s2h + j0 + 16 + 8 * half + 4);
        const v4i ad0 = *(const v4i*)(&sm_adj[buf][row * ADJ_STRIDE + 8 * half]);
        const v4i ad1 = *(const v4i*)(&sm_adj[buf][row * ADJ_STRIDE + 8 * half + 4]);
        const v4i ad2 = *(const v4i*)(&sm_adj[buf][row * ADJ_STRIDE + 16 + 8 * half]);
        const v4i ad3 = *(const v4i*)(&sm_adj[buf][row * ADJ_STRIDE + 16 + 8 * half + 4]);

        float ev[16];
        float lmax = -__builtin_inff();
        #pragma unroll
        for (int e = 0; e < 4; ++e) {
            float sa = s1v + s2v0[e]; sa = (sa >= 0.0f) ? sa : ALPHA * sa;
            float sb = s1v + s2v1[e]; sb = (sb >= 0.0f) ? sb : ALPHA * sb;
            float sc = s1v + s2v2[e]; sc = (sc >= 0.0f) ? sc : ALPHA * sc;
            float sd = s1v + s2v3[e]; sd = (sd >= 0.0f) ? sd : ALPHA * sd;
            ev[e]      = (ad0[e] > 0) ? sa : NEG_BIG;
            ev[e + 4]  = (ad1[e] > 0) ? sb : NEG_BIG;
            ev[e + 8]  = (ad2[e] > 0) ? sc : NEG_BIG;
            ev[e + 12] = (ad3[e] > 0) ? sd : NEG_BIG;
            lmax = fmaxf(lmax, fmaxf(fmaxf(ev[e], ev[e + 4]),
                                     fmaxf(ev[e + 8], ev[e + 12])));
        }
        // partner lane (lane^16) holds the other K-half of the same row
        const float cmax  = fmaxf(lmax, __shfl_xor(lmax, 16, 32));
        const float m_new = fmaxf(m_run, cmax);
        const float scale = __expf(m_run - m_new);

        v16bf P;
        float lsum = 0.0f;
        #pragma unroll
        for (int e = 0; e < 16; ++e) {
            const float p = __expf(ev[e] - m_new);
            lsum += p;
            P[e] = f2bf(p);
        }
        lsum += __shfl_xor(lsum, 16, 32);
        l_run = l_run * scale + lsum;
        m_run = m_new;

        // broadcast per-row rescale factors: row r's scale lives in lane r
        float scl[8];
        #pragma unroll
        for (int r = 0; r < 8; ++r) scl[r] = __shfl(scale, r + 8 * half, 32);

        #pragma unroll
        for (int t = 0; t < 4; ++t) {
            #pragma unroll
            for (int r = 0; r < 8; ++r) acc[t][r] *= scl[r];
            const v16bf b = ld_bfrag(
                WhbT + (long)(h * HD + 16 * t + row) * N_NODES + j0 + 8 * half);
            acc[t] = __builtin_amdgcn_wmma_f32_16x16x32_bf16(
                false, P, false, b, (short)0, acc[t], false, false);
        }
        __syncthreads();   // tile fully consumed before it can be overwritten
    }

    float invl[8];
    #pragma unroll
    for (int r = 0; r < 8; ++r)
        invl[r] = 1.0f / __shfl(l_run, r + 8 * half, 32);

    #pragma unroll
    for (int t = 0; t < 4; ++t)
        #pragma unroll
        for (int r = 0; r < 8; ++r) {
            float v = acc[t][r] * invl[r];
            v = (v > 0.0f) ? v : (__expf(v) - 1.0f);    // ELU
            out[(long)(i0 + r + 8 * half) * NF + h * HD + 16 * t + row] = v;
        }
}

// ------------------------------------------------------------------ launch
extern "C" void kernel_launch(void* const* d_in, const int* in_sizes, int n_in,
                              void* d_out, int out_size, void* d_ws, size_t ws_size,
                              hipStream_t stream) {
    const float* h   = (const float*)d_in[0];   // 4096*512 f32
    const int*   adj = (const int*)d_in[1];     // 4096*4096 i32
    const float* W   = (const float*)d_in[2];   // 512*512 f32
    const float* a   = (const float*)d_in[3];   // 1*8*128 f32
    float*       out = (float*)d_out;           // 4096*512 f32

    char* ws = (char*)d_ws;
    unsigned short* hb   = (unsigned short*)(ws + 0);          // 4 MB  bf16 h
    unsigned short* WbT  = (unsigned short*)(ws + 4194304);    // 512KB bf16 W^T
    float*          Wh   = (float*)(ws + 4718592);             // 8 MB  f32 Wh
    unsigned short* WhbT = (unsigned short*)(ws + 13107200);   // 4 MB  bf16 Wh^T
    float*          s1   = (float*)(ws + 17301504);            // 128KB
    float*          s2   = (float*)(ws + 17432576);            // 128KB

    gat_cvt_bf16<<<dim3(2048), 256, 0, stream>>>(h, hb, N_NODES * IN_F);
    gat_cvt_bf16_T<<<dim3(512), 256, 0, stream>>>(W, WbT, IN_F, NF);
    gat_wh_gemm<<<dim3(N_NODES / 16, NF / 64), 32, 0, stream>>>(hb, WbT, Wh, WhbT);
    gat_scores<<<dim3(N_NODES, NH), 32, 0, stream>>>(Wh, a, s1, s2);
    gat_attn<<<dim3(N_NODES / 16), 256, 0, stream>>>(s1, s2, WhbT, adj, out);
}
